// get_model_74251394613716
// MI455X (gfx1250) — matrix operand order repaired and checked by reading, not compiled, not run on hardware
//
#include <hip/hip_runtime.h>
#include <hip/hip_bf16.h>
#include <stddef.h>

// ---------------------------------------------------------------------------
// PointNet++ SSG fused pipeline for MI455X (gfx1250, wave32, WMMA).
// Dominant compute: per-point MLP 1024->512->256->32 over 65536 points
// (~86 GFLOP) -> v_wmma_f32_16x16x32_f16. Fully-unrolled K loops (scheduler
// pipelines loads, no register rotation), 64-point M tile with 2 A-subtiles
// per B fragment to halve weight traffic from L2.
// ---------------------------------------------------------------------------

#define BATCH 8
#define NPTS  8192
#define NCENT 36
#define NSAMP 36
#define MTILE 64

typedef _Float16 v16h __attribute__((ext_vector_type(16)));
typedef _Float16 v8h  __attribute__((ext_vector_type(8)));
typedef float    v8f  __attribute__((ext_vector_type(8)));
typedef float    v4f  __attribute__((ext_vector_type(4)));

// ---------------------------------------------------------------------------
// Weight repack: W[k][n] (f32, row-major KxN) * gamma[n]/sqrt(1+eps) -> f16,
// laid out as WMMA B fragments: frag = nt*KT + kt, within frag: lane*16+j,
// where col = nt*16 + (lane&15), k = kt*32 + (lane>>4)*8 + (j&7) + (j>>3)*16.
// Each lane then loads its 16 halves with one contiguous 32B read.
// ---------------------------------------------------------------------------
__global__ void repack_kernel(const float* __restrict__ w,
                              const float* __restrict__ gamma,
                              _Float16* __restrict__ dst, int K, int Nout) {
    int e = blockIdx.x * 256 + threadIdx.x;
    int total = K * Nout;
    if (e >= total) return;
    int KT   = K >> 5;
    int frag = e >> 9;         // /512
    int slot = e & 511;
    int lane = slot >> 4;
    int j    = slot & 15;
    int nt   = frag / KT;
    int kt   = frag % KT;
    int col  = nt * 16 + (lane & 15);
    int k    = kt * 32 + ((lane >> 4) * 8) + (j & 7) + ((j >> 3) * 16);
    float s  = gamma ? gamma[col] * rsqrtf(1.0f + 1e-5f) : 1.0f;
    dst[e] = (_Float16)(w[(size_t)k * Nout + col] * s);
}

// ---------------------------------------------------------------------------
// Farthest point sampling: one block per batch. dist[] lives in LDS.
// Matches jax.lax.scan semantics: first emitted index is 0.
// ---------------------------------------------------------------------------
__global__ __launch_bounds__(256) void fps_kernel(const float* __restrict__ xyz,
                                                  int* __restrict__ fps_idx,
                                                  float* __restrict__ new_xyz) {
    int b = blockIdx.x;
    const float* xb = xyz + (size_t)b * 6 * NPTS;
    __shared__ float dist[NPTS];
    __shared__ float sval[256];
    __shared__ int   sidx[256];
    __shared__ int   s_far;
    int t = threadIdx.x;
    for (int n = t; n < NPTS; n += 256) dist[n] = 1e10f;
    if (t == 0) s_far = 0;
    __syncthreads();
    for (int i = 0; i < NCENT; ++i) {
        int far = s_far;
        float cx0 = xb[far], cy0 = xb[NPTS + far], cz0 = xb[2 * NPTS + far];
        if (t == 0) {
            fps_idx[b * NCENT + i] = far;
            float* nz = new_xyz + (size_t)(b * NCENT + i) * 3;
            nz[0] = cx0; nz[1] = cy0; nz[2] = cz0;
        }
        float bv = -1.0f; int bi = 0;
        for (int n = t; n < NPTS; n += 256) {
            float dx = xb[n] - cx0, dy = xb[NPTS + n] - cy0, dz = xb[2 * NPTS + n] - cz0;
            float d = dx * dx + dy * dy + dz * dz;
            float dd = dist[n];
            if (d < dd) { dd = d; dist[n] = d; }
            if (dd > bv) { bv = dd; bi = n; }
        }
        sval[t] = bv; sidx[t] = bi;
        __syncthreads();
        for (int s = 128; s > 0; s >>= 1) {
            if (t < s) {
                float ov = sval[t + s]; int oi = sidx[t + s];
                if (ov > sval[t] || (ov == sval[t] && oi < sidx[t])) {
                    sval[t] = ov; sidx[t] = oi;
                }
            }
            __syncthreads();
        }
        if (t == 0) s_far = sidx[0];
        __syncthreads();
    }
}

// ---------------------------------------------------------------------------
// Ball query + grouped 9-ch features + 2-layer MLP(BN+ReLU) + max over 36
// samples, written into l1_points[b][center][coloff..coloff+C2).
// One block per (batch, center). Tiny compute -> plain VALU is fine.
// ---------------------------------------------------------------------------
__global__ __launch_bounds__(256) void branch_kernel(
    const float* __restrict__ xyz, const float* __restrict__ new_xyz,
    const float* __restrict__ w1, const float* __restrict__ g1,
    const float* __restrict__ b1, int C1,
    const float* __restrict__ w2, const float* __restrict__ g2,
    const float* __restrict__ b2, int C2,
    float r2, float* __restrict__ l1, int coloff) {
    int blk = blockIdx.x;
    int b = blk / NCENT, ci = blk % NCENT;
    const float* xb = xyz + (size_t)b * 6 * NPTS;
    __shared__ int   flags[256];
    __shared__ int   nidx[NSAMP];
    __shared__ int   s_cnt;
    __shared__ float grp[NSAMP * 9];
    __shared__ float h1s[NSAMP * 128];
    int t = threadIdx.x;
    const float* cz = new_xyz + (size_t)(b * NCENT + ci) * 3;
    float c0 = cz[0], c1v = cz[1], c2v = cz[2];
    if (t == 0) s_cnt = 0;
    __syncthreads();
    for (int base = 0; base < NPTS; base += 256) {
        int n = base + t;
        float dx = xb[n] - c0, dy = xb[NPTS + n] - c1v, dz = xb[2 * NPTS + n] - c2v;
        flags[t] = (dx * dx + dy * dy + dz * dz <= r2) ? 1 : 0;
        __syncthreads();
        if (t == 0) {
            int c = s_cnt;
            for (int i = 0; i < 256 && c < NSAMP; ++i)
                if (flags[i]) nidx[c++] = base + i;
            s_cnt = c;
        }
        __syncthreads();
        if (s_cnt >= NSAMP) break;
        __syncthreads();
    }
    if (t == 0 && s_cnt == 0) { nidx[0] = NPTS - 1; s_cnt = 1; }
    __syncthreads();
    int cnt = s_cnt;
    if (t < NSAMP && t >= cnt) nidx[t] = nidx[0];   // pad with first (ref semantics)
    __syncthreads();
    // grouped features: [pts 6ch, xyz-center 3ch]
    for (int e = t; e < NSAMP * 9; e += 256) {
        int s = e / 9, k = e % 9;
        int n = nidx[s];
        float v;
        if (k < 6) v = xb[(size_t)k * NPTS + n];
        else {
            float cc = (k == 6) ? c0 : ((k == 7) ? c1v : c2v);
            v = xb[(size_t)(k - 6) * NPTS + n] - cc;
        }
        grp[e] = v;
    }
    __syncthreads();
    float inv = rsqrtf(1.0f + 1e-5f);
    for (int e = t; e < NSAMP * C1; e += 256) {
        int s = e / C1, c = e % C1;
        float acc = 0.0f;
#pragma unroll
        for (int k = 0; k < 9; ++k) acc += grp[s * 9 + k] * w1[k * C1 + c];
        acc = acc * (g1[c] * inv) + b1[c];
        h1s[s * C1 + c] = acc > 0.0f ? acc : 0.0f;
    }
    __syncthreads();
    for (int c = t; c < C2; c += 256) {
        float sc = g2[c] * inv, be = b2[c];
        float m = -1e30f;
        for (int s = 0; s < NSAMP; ++s) {
            float acc = 0.0f;
            for (int k = 0; k < C1; ++k) acc += h1s[s * C1 + k] * w2[k * C2 + c];
            acc = acc * sc + be;
            acc = acc > 0.0f ? acc : 0.0f;
            if (acc > m) m = acc;
        }
        l1[((size_t)(b * NCENT + ci)) * 512 + coloff + c] = m;
    }
}

// ---------------------------------------------------------------------------
// Fused KNN-3 interpolation + FP MLP (1024->512->256) + conv (256->32) using
// v_wmma_f32_16x16x32_f16. One block = 64 points, 8 waves (256 threads).
// Wave mapping: rp = wave>>2 selects a 32-row pair, cq = wave&3 selects the
// output-column quarter. Each B fragment is multiplied against TWO A row
// tiles (rows rp*32+0..15 and rp*32+16..31) -> halves weight L2 traffic.
// K loops fully unrolled: scheduler pipelines ds/global loads under WMMAs.
// LDS reuse: the l1_points/KNN staging region is dead after the interp stage
// and is overlaid by the h1 activation tile.
// ---------------------------------------------------------------------------
__device__ __forceinline__ v16h load_a_frag(const _Float16* base, int stride,
                                            int row, int k0, int lhalf) {
    const v8h* plo = (const v8h*)(base + (size_t)row * stride + k0 + lhalf * 8);
    const v8h* phi = (const v8h*)(base + (size_t)row * stride + k0 + lhalf * 8 + 16);
    v8h lo = *plo, hi = *phi;
    v16h a;
#pragma unroll
    for (int i = 0; i < 8; ++i) { a[i] = lo[i]; a[i + 8] = hi[i]; }
    return a;
}

__global__ __launch_bounds__(256) void fused_fp_kernel(
    const float* __restrict__ tp, const float* __restrict__ g,
    const float* __restrict__ l1, const float* __restrict__ new_xyz,
    const _Float16* __restrict__ W1p, const _Float16* __restrict__ W2p,
    const _Float16* __restrict__ W3p,
    const float* __restrict__ beta1, const float* __restrict__ beta2,
    const float* __restrict__ convb, float* __restrict__ out) {
    extern __shared__ __align__(16) char smem[];
    // region 0 (65536 B): l1f(36864) + cx(448) + kidx(768) + kw(768) during
    // staging/interp; reused afterwards as h1b (64*512 f16 = 65536 B).
    _Float16* l1f  = (_Float16*)smem;
    float*    cx   = (float*)(smem + 36864);
    int*      kidx = (int*)(smem + 37312);
    float*    kw   = (float*)(smem + 38080);
    _Float16* h1b  = (_Float16*)smem;                    // overlay
    _Float16* xbuf = (_Float16*)(smem + 65536);          // 64*1024 f16 (131072)
    _Float16* h2b  = (_Float16*)(smem + 65536 + 131072); // 64*256 f16  (32768)
    // total 229376 B (single workgroup owns the WGP's 320KB LDS pool)

    int b    = blockIdx.x / (NPTS / MTILE);
    int tile = blockIdx.x % (NPTS / MTILE);
    int t    = threadIdx.x;

    // ---- stage l1_points (f32 -> f16), centers; replicate g into xbuf ------
    const float* l1base = l1 + (size_t)b * NCENT * 512;
    for (int e = t; e < NCENT * 512; e += 256) l1f[e] = (_Float16)l1base[e];
    for (int e = t; e < NCENT * 3; e += 256)  cx[e] = new_xyz[(size_t)b * NCENT * 3 + e];
    {
        _Float16 g0  = (_Float16)g[b * 512 + t];
        _Float16 g1h = (_Float16)g[b * 512 + 256 + t];
#pragma unroll 4
        for (int m = 0; m < MTILE; ++m) {
            xbuf[m * 1024 + t]       = g0;
            xbuf[m * 1024 + 256 + t] = g1h;
        }
    }
    __syncthreads();

    // ---- KNN-3 weights (one thread per point) ------------------------------
    if (t < MTILE) {
        int n = tile * MTILE + t;
        const float* tpb = tp + (size_t)b * 3 * NPTS;
        float px = tpb[n], py = tpb[NPTS + n], pz = tpb[2 * NPTS + n];
        float d0 = 1e30f, d1 = 1e30f, d2 = 1e30f;
        int i0 = 0, i1 = 0, i2 = 0;
        for (int c = 0; c < NCENT; ++c) {
            float dx = px - cx[c * 3], dy = py - cx[c * 3 + 1], dz = pz - cx[c * 3 + 2];
            float d = dx * dx + dy * dy + dz * dz;
            if (d < d0)      { d2 = d1; i2 = i1; d1 = d0; i1 = i0; d0 = d; i0 = c; }
            else if (d < d1) { d2 = d1; i2 = i1; d1 = d; i1 = c; }
            else if (d < d2) { d2 = d; i2 = c; }
        }
        float w0 = 1.0f / (d0 + 1e-8f), w1v = 1.0f / (d1 + 1e-8f), w2v = 1.0f / (d2 + 1e-8f);
        float ws = w0 + w1v + w2v;
        kidx[t * 3] = i0; kidx[t * 3 + 1] = i1; kidx[t * 3 + 2] = i2;
        kw[t * 3] = w0 / ws; kw[t * 3 + 1] = w1v / ws; kw[t * 3 + 2] = w2v / ws;
    }
    __syncthreads();

    // ---- build interp half of x: xbuf[m][512+c] ----------------------------
    for (int e = t; e < MTILE * 512; e += 256) {
        int m = e >> 9, c = e & 511;
        float acc = kw[m * 3]     * (float)l1f[kidx[m * 3]     * 512 + c]
                  + kw[m * 3 + 1] * (float)l1f[kidx[m * 3 + 1] * 512 + c]
                  + kw[m * 3 + 2] * (float)l1f[kidx[m * 3 + 2] * 512 + c];
        xbuf[m * 1024 + 512 + c] = (_Float16)acc;
    }
    __syncthreads();   // l1f/cx/kidx/kw dead; region reused as h1b

    int wv = t >> 5, lane = t & 31;
    int rp = wv >> 2, cq = wv & 3;
    int lrow = lane & 15, lhalf = lane >> 4;
    int row0 = rp * 32 + lrow;         // A subtile 0
    int row1 = rp * 32 + 16 + lrow;    // A subtile 1

    // ---- layer 1: x(64x1024) @ W1(1024x512) + beta1, relu ------------------
    for (int i = 0; i < 8; ++i) {
        int nt = cq * 8 + i;
        const _Float16* wbase = W1p + (size_t)nt * 32 * 512 + lane * 16;
        __builtin_prefetch(wbase + 32 * 512, 0, 3);   // next ntile's weights
        v8f acc0 = {}, acc1 = {};
#pragma unroll
        for (int kt = 0; kt < 32; ++kt) {
            v16h bf = *(const v16h*)(wbase + (size_t)kt * 512);
            v16h a0 = load_a_frag(xbuf, 1024, row0, kt * 32, lhalf);
            v16h a1 = load_a_frag(xbuf, 1024, row1, kt * 32, lhalf);
            acc0 = __builtin_amdgcn_wmma_f32_16x16x32_f16(false, a0, false, bf,
                                                          (short)0, acc0, false, false);
            acc1 = __builtin_amdgcn_wmma_f32_16x16x32_f16(false, a1, false, bf,
                                                          (short)0, acc1, false, false);
        }
        int col = nt * 16 + lrow;
        float be = beta1[col];
#pragma unroll
        for (int v = 0; v < 8; ++v) {
            int r = rp * 32 + lhalf * 8 + v;
            float x0 = acc0[v] + be;
            float x1 = acc1[v] + be;
            h1b[r * 512 + col]        = (_Float16)(x0 > 0.0f ? x0 : 0.0f);
            h1b[(r + 16) * 512 + col] = (_Float16)(x1 > 0.0f ? x1 : 0.0f);
        }
    }
    __syncthreads();

    // ---- layer 2: h1(64x512) @ W2(512x256) + beta2, relu -------------------
    for (int i = 0; i < 4; ++i) {
        int nt = cq * 4 + i;
        const _Float16* wbase = W2p + (size_t)nt * 16 * 512 + lane * 16;
        __builtin_prefetch(wbase + 16 * 512, 0, 3);
        v8f acc0 = {}, acc1 = {};
#pragma unroll
        for (int kt = 0; kt < 16; ++kt) {
            v16h bf = *(const v16h*)(wbase + (size_t)kt * 512);
            v16h a0 = load_a_frag(h1b, 512, row0, kt * 32, lhalf);
            v16h a1 = load_a_frag(h1b, 512, row1, kt * 32, lhalf);
            acc0 = __builtin_amdgcn_wmma_f32_16x16x32_f16(false, a0, false, bf,
                                                          (short)0, acc0, false, false);
            acc1 = __builtin_amdgcn_wmma_f32_16x16x32_f16(false, a1, false, bf,
                                                          (short)0, acc1, false, false);
        }
        int col = nt * 16 + lrow;
        float be = beta2[col];
#pragma unroll
        for (int v = 0; v < 8; ++v) {
            int r = rp * 32 + lhalf * 8 + v;
            float x0 = acc0[v] + be;
            float x1 = acc1[v] + be;
            h2b[r * 256 + col]        = (_Float16)(x0 > 0.0f ? x0 : 0.0f);
            h2b[(r + 16) * 256 + col] = (_Float16)(x1 > 0.0f ? x1 : 0.0f);
        }
    }
    __syncthreads();

    // ---- layer 3: h2(64x256) @ W3(256x32) + conv_b, write (B,32,N) ---------
    if (wv < 4) {                       // wave-uniform: EXEC stays all-ones
        int rp3 = wv >> 1, nt = wv & 1;
        int r0 = rp3 * 32 + lrow;
        int r1 = rp3 * 32 + 16 + lrow;
        const _Float16* wbase = W3p + (size_t)nt * 8 * 512 + lane * 16;
        v8f acc0 = {}, acc1 = {};
#pragma unroll
        for (int kt = 0; kt < 8; ++kt) {
            v16h bf = *(const v16h*)(wbase + (size_t)kt * 512);
            v16h a0 = load_a_frag(h2b, 256, r0, kt * 32, lhalf);
            v16h a1 = load_a_frag(h2b, 256, r1, kt * 32, lhalf);
            acc0 = __builtin_amdgcn_wmma_f32_16x16x32_f16(false, a0, false, bf,
                                                          (short)0, acc0, false, false);
            acc1 = __builtin_amdgcn_wmma_f32_16x16x32_f16(false, a1, false, bf,
                                                          (short)0, acc1, false, false);
        }
        int col = nt * 16 + lrow;
        float be = convb[col];
        int nbase0 = tile * MTILE + rp3 * 32 + lhalf * 8;
        float* op0 = out + (size_t)b * 32 * NPTS + (size_t)col * NPTS + nbase0;
        float* op1 = op0 + 16;
        v4f lo0 = { acc0[0] + be, acc0[1] + be, acc0[2] + be, acc0[3] + be };
        v4f hi0 = { acc0[4] + be, acc0[5] + be, acc0[6] + be, acc0[7] + be };
        v4f lo1 = { acc1[0] + be, acc1[1] + be, acc1[2] + be, acc1[3] + be };
        v4f hi1 = { acc1[4] + be, acc1[5] + be, acc1[6] + be, acc1[7] + be };
        *(v4f*)op0 = lo0;
        *(v4f*)(op0 + 4) = hi0;
        *(v4f*)op1 = lo1;
        *(v4f*)(op1 + 4) = hi1;
    }
}

// ---------------------------------------------------------------------------
// Host orchestration. Input order = jax pytree flatten of setup_inputs():
// 0 xyz, 1 g, 2 tp, then params dict (sorted keys): conv_b, conv_w,
// fp[0]{beta,gamma,w}, fp[1]{beta,gamma,w}, sa1[branch][layer]{beta,gamma,w}.
// ---------------------------------------------------------------------------
extern "C" void kernel_launch(void* const* d_in, const int* in_sizes, int n_in,
                              void* d_out, int out_size, void* d_ws, size_t ws_size,
                              hipStream_t stream) {
    (void)in_sizes; (void)n_in; (void)out_size; (void)ws_size;
    const float* xyz    = (const float*)d_in[0];
    const float* g      = (const float*)d_in[1];
    const float* tp     = (const float*)d_in[2];
    const float* conv_b = (const float*)d_in[3];
    const float* conv_w = (const float*)d_in[4];
    const float* fp0_b  = (const float*)d_in[5];
    const float* fp0_g  = (const float*)d_in[6];
    const float* fp0_w  = (const float*)d_in[7];
    const float* fp1_b  = (const float*)d_in[8];
    const float* fp1_g  = (const float*)d_in[9];
    const float* fp1_w  = (const float*)d_in[10];
    // sa1 branches: [11..16], [17..22], [23..28]
    const float* s00_b = (const float*)d_in[11];
    const float* s00_g = (const float*)d_in[12];
    const float* s00_w = (const float*)d_in[13];
    const float* s01_b = (const float*)d_in[14];
    const float* s01_g = (const float*)d_in[15];
    const float* s01_w = (const float*)d_in[16];
    const float* s10_b = (const float*)d_in[17];
    const float* s10_g = (const float*)d_in[18];
    const float* s10_w = (const float*)d_in[19];
    const float* s11_b = (const float*)d_in[20];
    const float* s11_g = (const float*)d_in[21];
    const float* s11_w = (const float*)d_in[22];
    const float* s20_b = (const float*)d_in[23];
    const float* s20_g = (const float*)d_in[24];
    const float* s20_w = (const float*)d_in[25];
    const float* s21_b = (const float*)d_in[26];
    const float* s21_g = (const float*)d_in[27];
    const float* s21_w = (const float*)d_in[28];

    char* ws = (char*)d_ws;
    size_t off = 0;
    auto take = [&](size_t bytes) -> char* {
        char* p = ws + off;
        off = (off + bytes + 255) & ~(size_t)255;
        return p;
    };
    int*      fps_i  = (int*)take((size_t)BATCH * NCENT * 4);
    float*    nx     = (float*)take((size_t)BATCH * NCENT * 3 * 4);
    float*    l1     = (float*)take((size_t)BATCH * NCENT * 512 * 4);
    _Float16* W1p    = (_Float16*)take((size_t)1024 * 512 * 2);
    _Float16* W2p    = (_Float16*)take((size_t)512 * 256 * 2);
    _Float16* W3p    = (_Float16*)take((size_t)256 * 32 * 2);

    // weight repack (BN scale folded in)
    repack_kernel<<<(1024 * 512 + 255) / 256, 256, 0, stream>>>(fp0_w, fp0_g, W1p, 1024, 512);
    repack_kernel<<<(512 * 256 + 255) / 256, 256, 0, stream>>>(fp1_w, fp1_g, W2p, 512, 256);
    repack_kernel<<<(256 * 32 + 255) / 256, 256, 0, stream>>>(conv_w, nullptr, W3p, 256, 32);

    // farthest point sampling
    fps_kernel<<<BATCH, 256, 0, stream>>>(xyz, fps_i, nx);

    // SA1 branches -> l1_points columns [0:128), [128:384), [384:512)
    branch_kernel<<<BATCH * NCENT, 256, 0, stream>>>(
        xyz, nx, s00_w, s00_g, s00_b, 64, s01_w, s01_g, s01_b, 128,
        0.1f * 0.1f, l1, 0);
    branch_kernel<<<BATCH * NCENT, 256, 0, stream>>>(
        xyz, nx, s10_w, s10_g, s10_b, 128, s11_w, s11_g, s11_b, 256,
        0.2f * 0.2f, l1, 128);
    branch_kernel<<<BATCH * NCENT, 256, 0, stream>>>(
        xyz, nx, s20_w, s20_g, s20_b, 64, s21_w, s21_g, s21_b, 128,
        0.4f * 0.4f, l1, 384);

    // fused interpolation + FP MLP + output conv (WMMA)
    size_t smem = 229376;   // l1f/h1b overlay + xbuf + h2b
    fused_fp_kernel<<<BATCH * (NPTS / MTILE), 256, smem, stream>>>(
        tp, g, l1, nx, W1p, W2p, W3p, fp0_b, fp1_b, conv_b, (float*)d_out);
}
